// NodeSearch_55155970015741
// MI455X (gfx1250) — compile-verified, matrix operand order checked by reading, not run on hardware
//
#include <hip/hip_runtime.h>
#include <math.h>

#define G_    32
#define N_    2048
#define IN_   128
#define L_    64
#define KSUB_ 64

typedef __attribute__((ext_vector_type(16))) _Float16 v16h;
typedef __attribute__((ext_vector_type(8)))  float    v8f;

__device__ __forceinline__ float eluf(float x) { return x > 0.f ? x : (expf(x) - 1.f); }

// ---------------------------------------------------------------------------
// deg: inv_deg[row] = 1 / sum_j g[row][j]   (row = flat over G*N)
// ---------------------------------------------------------------------------
__global__ __launch_bounds__(256) void deg_kernel(const float* __restrict__ g,
                                                  float* __restrict__ inv_deg) {
  const int row  = blockIdx.x * 8 + (threadIdx.x >> 5);
  const int lane = threadIdx.x & 31;
  const float4* gr = (const float4*)(g + (size_t)row * N_);
  float s = 0.f;
  #pragma unroll 4
  for (int j = lane; j < N_ / 4; j += 32) { float4 v = gr[j]; s += v.x + v.y + v.z + v.w; }
  #pragma unroll
  for (int off = 16; off; off >>= 1) s += __shfl_down(s, off, 32);
  if (lane == 0) inv_deg[row] = 1.f / s;
}

// ---------------------------------------------------------------------------
// Generic WMMA GEMM:  Out[M x 64] = act( A(f32->f16) @ Bfrag(f32->f16) )
//   mode 0: out = acc * outRowScale[row]              (xs = inv_deg .* (x@W))
//   mode 1: out = elu(acc + bias[col])                (hn)
//   mode 2: out = acc * outScalarArr[gi], A rows gathered by gidx (sub_reps)
// B rows are optionally shifted (-bShiftArr[gi]) and scaled (*bScale[row]).
// B fragments stored in LDS in native WMMA layout: per (ktile,ntile,lane)
// 16 halves so the inner loop is pure ds_load_b128 + v_wmma.
// ---------------------------------------------------------------------------
__global__ __launch_bounds__(256) void gemm64_kernel(
    const float* __restrict__ A, int lda, unsigned long long aStride,
    const int*   __restrict__ gidx,
    const float* __restrict__ B, unsigned long long bStride,
    const float* __restrict__ bScaleBase, unsigned long long bScaleStride,
    const float* __restrict__ bShiftArr,
    const float* __restrict__ outRowScaleBase, unsigned long long orsStride,
    const float* __restrict__ bias,
    const float* __restrict__ outScalarArr,
    float* __restrict__ Out, unsigned long long outStride,
    int M, int K, int mode)
{
  extern __shared__ _Float16 lfrag[];   // K*64 halves
  const int gi  = blockIdx.y;
  const int tid = threadIdx.x;
  const float* Ag = A + (unsigned long long)gi * aStride;
  const float* Bg = B + (unsigned long long)gi * bStride;
  const float  bShift = bShiftArr ? bShiftArr[gi] : 0.f;
  const float* bScale = bScaleBase ? (bScaleBase + (unsigned long long)gi * bScaleStride) : nullptr;

  // Build B fragments (WMMA 16-bit B layout: lane = half*16 + col,
  // slot j: K = ktile*32 + half*8 + j (j<8) or + 16 + half*8 + (j-8))
  for (int e = tid; e < K * 64; e += 256) {
    const int kk = e >> 6, n = e & 63;
    float v = Bg[(size_t)kk * 64 + n] - bShift;
    if (bScale) v *= bScale[kk];
    const int kt = kk >> 5, krem = kk & 31;
    const int hi   = (krem >> 3) & 1;
    const int slot = (krem & 7) | ((krem >> 4) << 3);
    const int lane = (hi << 4) | (n & 15);
    lfrag[(((size_t)(kt << 2) + (n >> 4)) * 32 + lane) * 16 + slot] = (_Float16)v;
  }
  __syncthreads();

  const int wave = tid >> 5, lane = tid & 31;
  const int row0 = blockIdx.x * 128 + wave * 16;
  if (row0 >= M) return;
  const int m = lane & 15, half = lane >> 4;
  int ar = row0 + m;
  if (gidx) ar = gidx[(size_t)gi * M + ar];
  const float* arow = Ag + (size_t)ar * lda;

  v8f acc0 = {}, acc1 = {}, acc2 = {}, acc3 = {};
  const v16h* bf = (const v16h*)lfrag;
  const int nkt = K >> 5;
  for (int kt = 0; kt < nkt; ++kt) {
    const int kb = (kt << 5) + (half << 3);
    float4 a0 = *(const float4*)(arow + kb);
    float4 a1 = *(const float4*)(arow + kb + 4);
    float4 a2 = *(const float4*)(arow + kb + 16);
    float4 a3 = *(const float4*)(arow + kb + 20);
    if (kt + 1 < nkt) __builtin_prefetch(arow + kb + 32);   // global_prefetch_b8
    v16h a;
    a[0]  = (_Float16)a0.x; a[1]  = (_Float16)a0.y; a[2]  = (_Float16)a0.z; a[3]  = (_Float16)a0.w;
    a[4]  = (_Float16)a1.x; a[5]  = (_Float16)a1.y; a[6]  = (_Float16)a1.z; a[7]  = (_Float16)a1.w;
    a[8]  = (_Float16)a2.x; a[9]  = (_Float16)a2.y; a[10] = (_Float16)a2.z; a[11] = (_Float16)a2.w;
    a[12] = (_Float16)a3.x; a[13] = (_Float16)a3.y; a[14] = (_Float16)a3.z; a[15] = (_Float16)a3.w;
    const int fb = (kt << 2) * 32 + lane;
    v16h b0 = bf[fb], b1 = bf[fb + 32], b2 = bf[fb + 64], b3 = bf[fb + 96];
    acc0 = __builtin_amdgcn_wmma_f32_16x16x32_f16(false, a, false, b0, (short)0, acc0, false, false);
    acc1 = __builtin_amdgcn_wmma_f32_16x16x32_f16(false, a, false, b1, (short)0, acc1, false, false);
    acc2 = __builtin_amdgcn_wmma_f32_16x16x32_f16(false, a, false, b2, (short)0, acc2, false, false);
    acc3 = __builtin_amdgcn_wmma_f32_16x16x32_f16(false, a, false, b3, (short)0, acc3, false, false);
  }

  // Epilogue. C/D layout: VGPR r -> row (half*8 + r), col = (lane&15) + nt*16
  const int col0 = lane & 15, rb = half << 3;
  const float sOut = (mode == 2 && outScalarArr) ? outScalarArr[gi] : 1.f;
  const float* ors = outRowScaleBase ? (outRowScaleBase + (unsigned long long)gi * orsStride) : nullptr;
  float* outG = Out + (unsigned long long)gi * outStride;
  #pragma unroll
  for (int r = 0; r < 8; ++r) {
    const int row = row0 + rb + r;
    const float rs = ors ? ors[row] : 1.f;
    float v[4] = {acc0[r], acc1[r], acc2[r], acc3[r]};
    #pragma unroll
    for (int nt = 0; nt < 4; ++nt) {
      const int col = col0 + (nt << 4);
      float xv = v[nt];
      if (mode == 1)      xv = eluf(xv + bias[col]);
      else if (mode == 0) xv *= rs;
      else                xv *= sOut;
      outG[(size_t)row * 64 + col] = xv;
    }
  }
}

// ---------------------------------------------------------------------------
// LN stats + raw scores + normalized column means, one block per graph
// ---------------------------------------------------------------------------
__global__ __launch_bounds__(256) void stats_kernel(
    const float* __restrict__ hn, const float* __restrict__ p_sel,
    float* __restrict__ scores, float* __restrict__ muArr,
    float* __restrict__ sArr, float* __restrict__ gmean)
{
  const int gi = blockIdx.x, tid = threadIdx.x;
  __shared__ float cs[64];
  __shared__ float rsum[256], rsq[256];
  __shared__ float sp[64];
  __shared__ float smu, ss;
  if (tid < 64) { cs[tid] = 0.f; sp[tid] = p_sel[tid]; }
  __syncthreads();
  const float* H = hn + (size_t)gi * N_ * 64;
  float ls = 0.f, lq = 0.f;
  for (int e = tid; e < N_ * 64; e += 256) { float v = H[e]; ls += v; lq += v * v; }
  atomicAdd(&cs[tid & 63], ls);          // stride 256 => fixed column per thread
  rsum[tid] = ls; rsq[tid] = lq;
  __syncthreads();
  for (int off = 128; off; off >>= 1) {
    if (tid < off) { rsum[tid] += rsum[tid + off]; rsq[tid] += rsq[tid + off]; }
    __syncthreads();
  }
  if (tid == 0) {
    const float inv = 1.f / (float)(N_ * 64);
    float mu  = rsum[0] * inv;
    float var = rsq[0] * inv - mu * mu;
    float s   = rsqrtf(var + 1e-5f);
    smu = mu; ss = s; muArr[gi] = mu; sArr[gi] = s;
  }
  __syncthreads();
  for (int row = tid; row < N_; row += 256) {
    const float* hr = H + (size_t)row * 64;
    float sc = 0.f;
    #pragma unroll 16
    for (int d = 0; d < 64; ++d) sc = fmaf(hr[d], sp[d], sc);
    scores[(size_t)gi * N_ + row] = sc;  // ranking invariant under LN affine
  }
  if (tid < 64) gmean[gi * 64 + tid] = ss * (cs[tid] * (1.f / N_) - smu);
}

// ---------------------------------------------------------------------------
// Sorted top-64 per graph (iterative argmax, ties -> smaller index)
// ---------------------------------------------------------------------------
__global__ __launch_bounds__(256) void topk_kernel(const float* __restrict__ scores,
                                                   int* __restrict__ idx) {
  const int gi = blockIdx.x, tid = threadIdx.x;
  const int wave = tid >> 5, lane = tid & 31;
  __shared__ float sv[N_];
  __shared__ float wv[8]; __shared__ int wi[8];
  for (int e = tid; e < N_; e += 256) sv[e] = scores[(size_t)gi * N_ + e];
  __syncthreads();
  for (int it = 0; it < KSUB_; ++it) {
    float bv = -3.4e38f; int bi = N_;
    for (int e = tid; e < N_; e += 256) {
      float v = sv[e];
      if (v > bv || (v == bv && e < bi)) { bv = v; bi = e; }
    }
    for (int off = 16; off; off >>= 1) {
      float ov = __shfl_down(bv, off, 32);
      int   oi = __shfl_down(bi, off, 32);
      if (ov > bv || (ov == bv && oi < bi)) { bv = ov; bi = oi; }
    }
    if (lane == 0) { wv[wave] = bv; wi[wave] = bi; }
    __syncthreads();
    if (tid == 0) {
      float fv = wv[0]; int fi = wi[0];
      for (int w = 1; w < 8; ++w)
        if (wv[w] > fv || (wv[w] == fv && wi[w] < fi)) { fv = wv[w]; fi = wi[w]; }
      idx[gi * KSUB_ + it] = fi;
      sv[fi] = -3.4e38f;
    }
    __syncthreads();
  }
}

// ---------------------------------------------------------------------------
// ops -> conv1d(k=2) -> relu -> maxpool -> alpha mixture, per graph
// ---------------------------------------------------------------------------
__global__ __launch_bounds__(256) void conv_kernel(
    const float* __restrict__ subreps, const float* __restrict__ gmean,
    const float* __restrict__ conv_k, const float* __restrict__ conv_b,
    const float* __restrict__ log_alphas, float* __restrict__ Bmat)
{
  const int gi = blockIdx.x, tid = threadIdx.x;
  __shared__ float sub[KSUB_ * 64];     // 16 KB
  __shared__ float ck[2 * 64 * 64];     // 32 KB
  __shared__ float gm[64], cb[64];
  __shared__ float feats[3 * 64];
  for (int e = tid; e < KSUB_ * 64; e += 256) sub[e] = subreps[(size_t)gi * KSUB_ * 64 + e];
  for (int e = tid; e < 2 * 64 * 64; e += 256) ck[e] = conv_k[e];
  if (tid < 64) { gm[tid] = gmean[gi * 64 + tid]; cb[tid] = conv_b[tid]; }
  __syncthreads();
  float la0 = log_alphas[0], la1 = log_alphas[1], la2 = log_alphas[2];
  float mx = fmaxf(la0, fmaxf(la1, la2));
  float e0 = expf(la0 - mx), e1 = expf(la1 - mx), e2 = expf(la2 - mx);
  float inv = 1.f / (e0 + e1 + e2);
  float al0 = e0 * inv, al1 = e1 * inv, al2 = e2 * inv;
  if (tid < 192) {
    const int o = tid >> 6, d = tid & 63;
    float fm = -3.4e38f;
    for (int t = 0; t < KSUB_ - 1; ++t) {
      float acc = cb[d];
      for (int l = 0; l < 64; ++l) {
        float v0 = sub[t * 64 + l], v1 = sub[(t + 1) * 64 + l];
        if (o == 1)      { v0 = eluf(v0); v1 = eluf(v1); }
        else if (o == 2) { v0 += gm[l];   v1 += gm[l]; }
        acc = fmaf(v0, ck[l * 64 + d], acc);
        acc = fmaf(v1, ck[4096 + l * 64 + d], acc);
      }
      fm = fmaxf(fm, acc);
    }
    feats[o * 64 + d] = fmaxf(fm, 0.f);   // max of relu == relu of max
  }
  __syncthreads();
  if (tid < 64)
    Bmat[gi * 64 + tid] = al0 * feats[tid] + al1 * feats[64 + tid] + al2 * feats[128 + tid];
}

// ---------------------------------------------------------------------------
// Discriminator logits: out[0:32] = sum(B*wc), out[32:64] = sum(roll(B,1)*wc)
// ---------------------------------------------------------------------------
__global__ __launch_bounds__(256) void final_kernel(
    const float* __restrict__ original, const float* __restrict__ W_disc,
    const float* __restrict__ Bmat, float* __restrict__ out)
{
  const int tid = threadIdx.x;
  __shared__ float wc[G_ * 64];
  for (int e = tid; e < G_ * 64; e += 256) {
    const int i = e >> 6, d = e & 63;
    float s = 0.f;
    for (int l = 0; l < 64; ++l) s = fmaf(original[i * 64 + l], W_disc[l * 64 + d], s);
    wc[e] = s;
  }
  __syncthreads();
  if (tid < 64) {
    const int half = tid >> 5, i = tid & 31;
    const int src = half ? ((i + G_ - 1) & (G_ - 1)) : i;   // roll(B, 1)
    float s = 0.f;
    for (int d = 0; d < 64; ++d) s = fmaf(Bmat[src * 64 + d], wc[i * 64 + d], s);
    out[half * 32 + i] = s;
  }
}

// ---------------------------------------------------------------------------
extern "C" void kernel_launch(void* const* d_in, const int* in_sizes, int n_in,
                              void* d_out, int out_size, void* d_ws, size_t ws_size,
                              hipStream_t stream)
{
  (void)in_sizes; (void)n_in; (void)out_size; (void)ws_size;
  const float* original   = (const float*)d_in[0];
  const float* x          = (const float*)d_in[1];
  const float* g          = (const float*)d_in[2];
  const float* W_gcn      = (const float*)d_in[3];
  const float* b_gcn      = (const float*)d_in[4];
  const float* p_select   = (const float*)d_in[5];
  const float* conv_k     = (const float*)d_in[6];
  const float* conv_b     = (const float*)d_in[7];
  const float* W_disc     = (const float*)d_in[8];
  const float* log_alphas = (const float*)d_in[9];
  float* out = (float*)d_out;

  // workspace carve (~34 MB)
  float* inv_deg = (float*)d_ws;                        // G*N
  float* xs      = inv_deg + (size_t)G_ * N_;           // G*N*64
  float* hn      = xs + (size_t)G_ * N_ * 64;           // G*N*64
  float* scores  = hn + (size_t)G_ * N_ * 64;           // G*N
  float* muArr   = scores + (size_t)G_ * N_;            // G
  float* sArr    = muArr + G_;                          // G
  float* gmean   = sArr + G_;                           // G*64
  int*   idx     = (int*)(gmean + G_ * 64);             // G*KSUB
  float* subrep  = (float*)(idx + G_ * KSUB_);          // G*KSUB*64
  float* Bmat    = subrep + (size_t)G_ * KSUB_ * 64;    // G*64

  deg_kernel<<<G_ * N_ / 8, 256, 0, stream>>>(g, inv_deg);

  // xs[j,:] = inv_deg[j] * (x[j,:] @ W_gcn)          (mode 0, K=128)
  gemm64_kernel<<<dim3(N_ / 128, G_), 256, (size_t)IN_ * 64 * 2, stream>>>(
      x, IN_, (unsigned long long)N_ * IN_, nullptr,
      W_gcn, 0ull, nullptr, 0ull, nullptr,
      inv_deg, (unsigned long long)N_, nullptr, nullptr,
      xs, (unsigned long long)N_ * 64, N_, IN_, 0);

  // hn = elu(g @ xs + b_gcn)                          (mode 1, K=2048)
  gemm64_kernel<<<dim3(N_ / 128, G_), 256, (size_t)N_ * 64 * 2, stream>>>(
      g, N_, (unsigned long long)N_ * N_, nullptr,
      xs, (unsigned long long)N_ * 64, nullptr, 0ull, nullptr,
      nullptr, 0ull, b_gcn, nullptr,
      hn, (unsigned long long)N_ * 64, N_, N_, 1);

  stats_kernel<<<G_, 256, 0, stream>>>(hn, p_select, scores, muArr, sArr, gmean);
  topk_kernel<<<G_, 256, 0, stream>>>(scores, idx);

  // sub_reps = s * (g[idx,:] @ (inv_deg .* (hn - mu)))   (mode 2, K=2048)
  gemm64_kernel<<<dim3(1, G_), 256, (size_t)N_ * 64 * 2, stream>>>(
      g, N_, (unsigned long long)N_ * N_, idx,
      hn, (unsigned long long)N_ * 64, inv_deg, (unsigned long long)N_, muArr,
      nullptr, 0ull, nullptr, sArr,
      subrep, (unsigned long long)KSUB_ * 64, KSUB_, N_, 2);

  conv_kernel<<<G_, 256, 0, stream>>>(subrep, gmean, conv_k, conv_b, log_alphas, Bmat);
  final_kernel<<<1, 256, 0, stream>>>(original, W_disc, Bmat, out);
}
